// StreamingSSMCell_83434034692784
// MI455X (gfx1250) — compile-verified, matrix operand order checked by reading, not run on hardware
//
#include <hip/hip_runtime.h>
#include <hip/hip_bf16.h>

// ---------------------------------------------------------------------------
// StreamingSSMCell for MI455X (gfx1250, wave32, WMMA + async global->LDS DMA).
//   xz  = x @ W_in^T + b_in                  (WMMA f16 GEMM, f32 accum)
//   fused: roll conv_buf, depthwise conv, silu, decay recurrence, gate
//   out = (h_new*silu(z)) @ W_out^T + b_out  (WMMA f16 GEMM, f32 accum)
// ---------------------------------------------------------------------------

typedef __attribute__((ext_vector_type(16))) _Float16 v16h;
typedef __attribute__((ext_vector_type(8)))  _Float16 v8h;
typedef __attribute__((ext_vector_type(8)))  float    v8f;

#define BM 128
#define BN 128
#define BK 32
#define LDT 40   // padded LDS k-stride in halves: 20-bank step -> conflict-free b128 reads

// ---------------------------------------------------------------------------
// f32 -> f16 conversion, 4 elements/thread (n must be divisible by 4)
// ---------------------------------------------------------------------------
__global__ void cvt_f32_to_f16(const float* __restrict__ src,
                               _Float16* __restrict__ dst, int n4) {
    int i = blockIdx.x * blockDim.x + threadIdx.x;
    if (i >= n4) return;
    const float4 v = reinterpret_cast<const float4*>(src)[i];
    _Float16 h0 = (_Float16)v.x, h1 = (_Float16)v.y,
             h2 = (_Float16)v.z, h3 = (_Float16)v.w;
    struct H4 { _Float16 a, b, c, d; };
    reinterpret_cast<H4*>(dst)[i] = H4{h0, h1, h2, h3};
}

// ---------------------------------------------------------------------------
// C[m][n] = sum_k A[m][k] * Bw[n][k] + bias[n]
// A: MxK f16 row-major.  Bw: NxK f16 row-major (i.e. B = Bw^T).  C: MxN f32.
// 256 threads = 8 waves; block tile 128x128; wave tile 64x32 (4x2 WMMA tiles);
// double-buffered LDS fed by GLOBAL_LOAD_ASYNC_TO_LDS_B128 (ASYNCcnt-tracked
// DMA, no VGPR staging), K stepped by 32.
// M,N divisible by 128; K divisible by 32 (holds for all uses here).
// ---------------------------------------------------------------------------
__global__ __launch_bounds__(256)
void wmma_gemm_nt(const _Float16* __restrict__ A,
                  const _Float16* __restrict__ Bw,
                  const float* __restrict__ bias,
                  float* __restrict__ C,
                  int M, int N, int K) {
    __shared__ __attribute__((aligned(16))) _Float16 Al[2][BM * LDT];
    __shared__ __attribute__((aligned(16))) _Float16 Bl[2][BN * LDT];

    const int tid  = threadIdx.x;
    const int m0   = blockIdx.y * BM;
    const int n0   = blockIdx.x * BN;

    // global -> LDS staging: each thread DMAs 16 halves (32B) of A and of B
    const int row  = tid >> 1;        // 0..127
    const int seg  = tid & 1;         // halves 0..15 / 16..31 of the k-chunk

    const unsigned long long Abase = (unsigned long long)(size_t)A;
    const unsigned long long Bbase = (unsigned long long)(size_t)Bw;
    const unsigned aoff = (unsigned)((((size_t)(m0 + row)) * K + seg * 16) * sizeof(_Float16));
    const unsigned boff = (unsigned)((((size_t)(n0 + row)) * K + seg * 16) * sizeof(_Float16));
    const unsigned ldsA0 = (unsigned)(size_t)(&Al[0][row * LDT + seg * 16]);
    const unsigned ldsB0 = (unsigned)(size_t)(&Bl[0][row * LDT + seg * 16]);

    const int wave = tid >> 5;        // 0..7
    const int lane = tid & 31;
    const int r    = lane & 15;       // row (A) / col (B) within 16x16 tile
    const int hlf  = lane >> 4;       // half-wave: K groups 0..7/16..23 vs 8..15/24..31
    const int wm   = wave & 1;        // 2 waves along M
    const int wn   = wave >> 1;       // 4 waves along N

    v8f acc[4][2] = {};               // 8 x (16x16 f32) accumulators

    // Issue the 4 async DMA b128 loads for one K-chunk (A + B, 32B each).
    // INST_OFFSET is added to BOTH the global and LDS addresses (ISA 08 §4.4),
    // so offset:16 covers the second 16B chunk on both sides.
    auto issue = [&](int kt, int buf) {
        const unsigned kb = (unsigned)(kt * BK * sizeof(_Float16));
        const unsigned gA = aoff + kb;
        const unsigned gB = boff + kb;
        const unsigned lA = ldsA0 + (unsigned)(buf * (BM * LDT * (int)sizeof(_Float16)));
        const unsigned lB = ldsB0 + (unsigned)(buf * (BN * LDT * (int)sizeof(_Float16)));
        asm volatile("global_load_async_to_lds_b128 %0, %1, %2"
                     :: "v"(lA), "v"(gA), "s"(Abase) : "memory");
        asm volatile("global_load_async_to_lds_b128 %0, %1, %2 offset:16"
                     :: "v"(lA), "v"(gA), "s"(Abase) : "memory");
        asm volatile("global_load_async_to_lds_b128 %0, %1, %2"
                     :: "v"(lB), "v"(gB), "s"(Bbase) : "memory");
        asm volatile("global_load_async_to_lds_b128 %0, %1, %2 offset:16"
                     :: "v"(lB), "v"(gB), "s"(Bbase) : "memory");
    };

    issue(0, 0);

    const int nk = K / BK;
    for (int kt = 0; kt < nk; ++kt) {
        // my tile-kt DMAs have landed in LDS
        asm volatile("s_wait_asynccnt 0x0" ::: "memory");
        // everyone's tile-kt DMAs landed; everyone finished reading buf (kt+1)&1
        __syncthreads();
        if (kt + 1 < nk) issue(kt + 1, (kt + 1) & 1);   // DMA next chunk, other buffer

        const int buf = kt & 1;
        v16h af[4], bf[2];
        #pragma unroll
        for (int mt = 0; mt < 4; ++mt) {
            const _Float16* p = &Al[buf][(wm * 64 + mt * 16 + r) * LDT + hlf * 8];
            ((v8h*)&af[mt])[0] = *(const v8h*)(p);        // K 0..7   (or 8..15)
            ((v8h*)&af[mt])[1] = *(const v8h*)(p + 16);   // K 16..23 (or 24..31)
        }
        #pragma unroll
        for (int nt = 0; nt < 2; ++nt) {
            const _Float16* p = &Bl[buf][(wn * 32 + nt * 16 + r) * LDT + hlf * 8];
            ((v8h*)&bf[nt])[0] = *(const v8h*)(p);
            ((v8h*)&bf[nt])[1] = *(const v8h*)(p + 16);
        }
        #pragma unroll
        for (int mt = 0; mt < 4; ++mt)
            #pragma unroll
            for (int nt = 0; nt < 2; ++nt)
                acc[mt][nt] = __builtin_amdgcn_wmma_f32_16x16x32_f16(
                    false, af[mt], false, bf[nt],
                    (short)0, acc[mt][nt], false, false);
    }

    // epilogue: C-layout = lane(r,hlf), VGPR j -> (M = hlf*8 + j, N = r)
    #pragma unroll
    for (int mt = 0; mt < 4; ++mt) {
        #pragma unroll
        for (int nt = 0; nt < 2; ++nt) {
            const int nn = n0 + wn * 32 + nt * 16 + r;
            const float bv = bias[nn];
            #pragma unroll
            for (int j = 0; j < 8; ++j) {
                const int mm = m0 + wm * 64 + mt * 16 + hlf * 8 + j;
                C[(size_t)mm * N + nn] = acc[mt][nt][j] + bv;
            }
        }
    }
}

// ---------------------------------------------------------------------------
// Fused elementwise stage: ring-buffer roll + depthwise conv + SiLU + decay
// recurrence + gate. Emits h_new and new_buf straight to d_out, and the f16
// GEMM2 input g = h_new * silu(z) to workspace.
// ---------------------------------------------------------------------------
__global__ __launch_bounds__(256)
void ssm_fuse(const float* __restrict__ xz,       // [B, 2D]
              const float* __restrict__ h,        // [B, D]
              const float* __restrict__ conv_buf, // [B, D, 4]
              const float* __restrict__ conv_w,   // [D, 4]
              const float* __restrict__ conv_b,   // [D]
              float* __restrict__ h_out,          // [B, D]
              float* __restrict__ buf_out,        // [B, D, 4]
              _Float16* __restrict__ g_out,       // [B, D]
              int D_) {
    const int idx = blockIdx.x * blockDim.x + threadIdx.x;   // b*D + d
    const int d = idx & (D_ - 1);
    const int b = idx >> 10;   // D = 1024

    const float4 cb = reinterpret_cast<const float4*>(conv_buf)[idx];
    const float4 w  = reinterpret_cast<const float4*>(conv_w)[d];
    const size_t xzrow = (size_t)b * 2 * D_;
    const float xin = xz[xzrow + d];
    const float zz  = xz[xzrow + D_ + d];

    // rolled buffer: [cb.y, cb.z, cb.w, xin]
    const float nb0 = cb.y, nb1 = cb.z, nb2 = cb.w, nb3 = xin;
    float co = nb0 * w.x + nb1 * w.y + nb2 * w.z + nb3 * w.w + conv_b[d];
    co = co / (1.0f + __expf(-co));                          // silu

    // decay[d] = exp(-exp(-linspace(log 10, log 2000, D)[d]))
    const float LOGMIN = 2.302585092994046f;
    const float LOGMAX = 7.600902459542082f;
    const float lt = LOGMIN + (LOGMAX - LOGMIN) * ((float)d / (float)(D_ - 1));
    const float decay = __expf(-__expf(-lt));

    const float hn = decay * h[idx] + (1.0f - decay) * co;
    h_out[idx] = hn;
    reinterpret_cast<float4*>(buf_out)[idx] = float4{nb0, nb1, nb2, nb3};

    const float sz = zz / (1.0f + __expf(-zz));              // silu(z)
    g_out[idx] = (_Float16)(hn * sz);
}

// ---------------------------------------------------------------------------
extern "C" void kernel_launch(void* const* d_in, const int* in_sizes, int n_in,
                              void* d_out, int out_size, void* d_ws, size_t ws_size,
                              hipStream_t stream) {
    (void)in_sizes; (void)n_in; (void)out_size; (void)ws_size;

    const float* x        = (const float*)d_in[0];
    const float* h        = (const float*)d_in[1];
    const float* conv_buf = (const float*)d_in[2];
    const float* W_in     = (const float*)d_in[3];
    const float* b_in     = (const float*)d_in[4];
    const float* conv_w   = (const float*)d_in[5];
    const float* conv_b   = (const float*)d_in[6];
    const float* W_out    = (const float*)d_in[7];
    const float* b_out    = (const float*)d_in[8];

    const int Bb = 8192, Dd = 1024;
    const size_t BD = (size_t)Bb * Dd;

    // workspace layout (halves then floats), ~102 MiB total
    _Float16* wWin  = (_Float16*)d_ws;            // 2048*1024 halves (4 MiB)
    _Float16* wWout = wWin  + (size_t)2048 * 1024;//            (2 MiB)
    _Float16* wX    = wWout + (size_t)1024 * 1024;// B*D halves (16 MiB)
    _Float16* wG    = wX    + BD;                 // B*D halves (16 MiB)
    float*    wXZ   = (float*)(wG + BD);          // B*2D floats (64 MiB)

    float* out     = (float*)d_out;               // [B, D]
    float* h_out   = out + BD;                    // [B, D]
    float* buf_out = h_out + BD;                  // [B, D, 4]

    // 1) f32 -> f16 conversions (weights + x)
    {
        int n4 = (2048 * 1024) / 4;
        cvt_f32_to_f16<<<(n4 + 255) / 256, 256, 0, stream>>>(W_in, wWin, n4);
        n4 = (1024 * 1024) / 4;
        cvt_f32_to_f16<<<(n4 + 255) / 256, 256, 0, stream>>>(W_out, wWout, n4);
        n4 = (int)(BD / 4);
        cvt_f32_to_f16<<<(n4 + 255) / 256, 256, 0, stream>>>(x, wX, n4);
    }

    // 2) xz = x @ W_in^T + b_in   (M=8192, N=2048, K=1024)
    {
        dim3 grid(2048 / BN, 8192 / BM);
        wmma_gemm_nt<<<grid, 256, 0, stream>>>(wX, wWin, b_in, wXZ, Bb, 2 * Dd, Dd);
    }

    // 3) fused elementwise stage
    ssm_fuse<<<(int)(BD / 256), 256, 0, stream>>>(wXZ, h, conv_buf, conv_w, conv_b,
                                                  h_out, buf_out, wG, Dd);

    // 4) out = g @ W_out^T + b_out   (M=8192, N=1024, K=1024)
    {
        dim3 grid(1024 / BN, 8192 / BM);
        wmma_gemm_nt<<<grid, 256, 0, stream>>>(wG, wWout, b_out, out, Bb, Dd, Dd);
    }
}
